// HKSABlock_11433202942765
// MI455X (gfx1250) — matrix operand order repaired and checked by reading, not compiled, not run on hardware
//
#include <hip/hip_runtime.h>
#include <stdint.h>

typedef __bf16 bf16_t;
typedef __bf16 v16bf __attribute__((ext_vector_type(16)));
typedef float  v8f   __attribute__((ext_vector_type(8)));
typedef unsigned int uint4v __attribute__((ext_vector_type(4)));
typedef unsigned int u32x4  __attribute__((ext_vector_type(4)));
typedef int          i32x4  __attribute__((ext_vector_type(4)));
typedef int          i32x8  __attribute__((ext_vector_type(8)));

#define WMMA_BF16(a, b, c) \
  __builtin_amdgcn_wmma_f32_16x16x32_bf16(false, (a), false, (b), (short)0, (c), false, false)

static __device__ __forceinline__ float red_max16(float v) {
#pragma unroll
  for (int m = 8; m >= 1; m >>= 1) v = fmaxf(v, __shfl_xor(v, m, 32));
  return v;
}
static __device__ __forceinline__ float red_sum16(float v) {
#pragma unroll
  for (int m = 8; m >= 1; m >>= 1) v += __shfl_xor(v, m, 32);
  return v;
}

// ---------------------------------------------------------------- cast f32->bf16
__global__ void cast_f32_bf16(const float* __restrict__ in, bf16_t* __restrict__ out, int n) {
  int i = blockIdx.x * blockDim.x + threadIdx.x;
  if (i < n) out[i] = (bf16_t)in[i];
}

// ---------------------------------------------------------------- RMSNorm (row/block)
__global__ void __launch_bounds__(256)
rmsnorm_bf16(const float* __restrict__ x, const float* __restrict__ w,
             bf16_t* __restrict__ out, int D) {
  __shared__ float red[8];
  const int row = blockIdx.x;
  const float* xr = x + (size_t)row * D;
  float s = 0.f;
  for (int c = threadIdx.x; c < D; c += 256) { float v = xr[c]; s += v * v; }
#pragma unroll
  for (int m = 16; m >= 1; m >>= 1) s += __shfl_xor(s, m, 32);
  const int wid = threadIdx.x >> 5, lid = threadIdx.x & 31;
  if (lid == 0) red[wid] = s;
  __syncthreads();
  if (threadIdx.x == 0) {
    float t = 0.f;
#pragma unroll
    for (int i = 0; i < 8; i++) t += red[i];
    red[0] = t;
  }
  __syncthreads();
  const float rms = rsqrtf(red[0] / (float)D + 1e-5f);
  bf16_t* orow = out + (size_t)row * D;
  for (int c = threadIdx.x; c < D; c += 256) orow[c] = (bf16_t)(xr[c] * rms * w[c]);
}

// ---------------------------------------------------------------- TDM L2 warm
// Pulls 32KB tiles of a weight matrix through the cache hierarchy with the
// Tensor Data Mover (D# built per CDNA5 ISA 8.3/8.4; 2-D, 8B elements).
__global__ void __launch_bounds__(32)
l2_warm_tdm(const void* __restrict__ src, int rowElems8, int rowsTotal, int tilesX) {
  __shared__ long long buf[4096];  // 32 KB destination
  const int tx = blockIdx.x % tilesX;
  const int ty = blockIdx.x / tilesX;
  if (threadIdx.x == 0) {
    const unsigned long long gaddr =
        (unsigned long long)(size_t)src +
        ((unsigned long long)ty * 16ull * (unsigned long long)rowElems8 +
         (unsigned long long)tx * 256ull) * 8ull;
    const unsigned lds = (unsigned)(size_t)(void*)buf;  // low 32 bits = LDS offset
    u32x4 g0;
    g0[0] = 1u;                                          // count=1, user descriptor
    g0[1] = lds;                                         // lds_addr
    g0[2] = (unsigned)(gaddr & 0xffffffffu);             // global_addr[31:0]
    g0[3] = (unsigned)((gaddr >> 32) & 0x01ffffffu) | (2u << 30);  // addr[56:32], type=2
    const unsigned dim0 = (unsigned)rowElems8;           // tensor_dim0 (8B units)
    const unsigned dim1 = (unsigned)rowsTotal;
    const unsigned long long s0 = (unsigned long long)rowElems8;  // dim0 stride
    i32x8 g1;
    g1[0] = (int)(3u << 16);                                        // data_size=8B
    g1[1] = (int)((dim0 & 0xffffu) << 16);                          // dim0[15:0]
    g1[2] = (int)(((dim0 >> 16) & 0xffffu) | ((dim1 & 0xffffu) << 16));
    g1[3] = (int)(((dim1 >> 16) & 0xffffu) | (256u << 16));         // tile_dim0=256
    g1[4] = (int)16;                                                // tile_dim1=16
    g1[5] = (int)(unsigned)(s0 & 0xffffffffu);                      // stride0[31:0]
    g1[6] = (int)(unsigned)((s0 >> 32) & 0xffffu);                  // stride0[47:32]
    g1[7] = 0;
    i32x4 z4 = {0, 0, 0, 0};
#if defined(__clang_major__) && (__clang_major__ >= 23)
    i32x8 z8 = {0, 0, 0, 0, 0, 0, 0, 0};
    __builtin_amdgcn_tensor_load_to_lds(g0, g1, z4, z4, z8, 0);
#else
    __builtin_amdgcn_tensor_load_to_lds(g0, g1, z4, z4, 0);
#endif
    __builtin_amdgcn_s_wait_tensorcnt(0);
  }
}

// ---------------------------------------------------------------- WMMA GEMM
// C[M,N] (f32, += residual) = A[M,K](bf16 rm) * B[K,N](bf16 rm)
// block 128x128, 8 waves (2x4), wave tile 64x32 (4x2 WMMA), K step 32,
// register-staged software pipeline, padded LDS (conflict-free b128 frags).
__global__ void __launch_bounds__(256)
gemm_bf16_f32(const bf16_t* __restrict__ A, const bf16_t* __restrict__ B,
              float* __restrict__ C, const float* __restrict__ Res,
              int M, int N, int K) {
  __shared__ bf16_t As[128][40];   // [m][k], padded
  __shared__ bf16_t BsT[128][40];  // [n][k], padded
  const int bm = blockIdx.y * 128;
  const int bn = blockIdx.x * 128;
  const int tid = threadIdx.x;
  const int wave = tid >> 5, lane = tid & 31;
  const int wm = wave >> 2, wn = wave & 3;
  const int lane16 = lane & 15, laneHi = (lane >> 4) & 1;

  v8f acc[4][2];
#pragma unroll
  for (int i = 0; i < 4; i++)
#pragma unroll
    for (int j = 0; j < 2; j++)
#pragma unroll
      for (int e = 0; e < 8; e++) acc[i][j][e] = 0.f;

  // staging coords: A chunks c,c+256 -> row=c>>2, col8=(c&3)*8
  //                 B chunks c,c+256 -> kk=c>>4, n8=(c&15)*8
  const int aR0 = tid >> 2, aC0 = (tid & 3) * 8;
  const int aR1 = (tid + 256) >> 2, aC1 = ((tid + 256) & 3) * 8;
  const int bK0 = tid >> 4, bN0 = (tid & 15) * 8;
  const int bK1 = (tid + 256) >> 4, bN1 = ((tid + 256) & 15) * 8;

  union U8 { uint4v q; bf16_t h[8]; };
  U8 a0, a1, b0, b1;
  a0.q = *(const uint4v*)(A + (size_t)(bm + aR0) * K + aC0);
  a1.q = *(const uint4v*)(A + (size_t)(bm + aR1) * K + aC1);
  b0.q = *(const uint4v*)(B + (size_t)bK0 * N + bn + bN0);
  b1.q = *(const uint4v*)(B + (size_t)bK1 * N + bn + bN1);

  for (int k0 = 0; k0 < K; k0 += 32) {
    // store staged regs to LDS (B transposed during store)
    *(uint4v*)&As[aR0][aC0] = a0.q;
    *(uint4v*)&As[aR1][aC1] = a1.q;
#pragma unroll
    for (int j = 0; j < 8; j++) BsT[bN0 + j][bK0] = b0.h[j];
#pragma unroll
    for (int j = 0; j < 8; j++) BsT[bN1 + j][bK1] = b1.h[j];
    __syncthreads();

    // issue next tile's global loads; they overlap the WMMA work below
    if (k0 + 32 < K) {
      const int kn = k0 + 32;
      a0.q = *(const uint4v*)(A + (size_t)(bm + aR0) * K + kn + aC0);
      a1.q = *(const uint4v*)(A + (size_t)(bm + aR1) * K + kn + aC1);
      b0.q = *(const uint4v*)(B + (size_t)(kn + bK0) * N + bn + bN0);
      b1.q = *(const uint4v*)(B + (size_t)(kn + bK1) * N + bn + bN1);
    }

    v16bf af[4], bfr[2];
#pragma unroll
    for (int i = 0; i < 4; i++) {
      const unsigned short* arow = (const unsigned short*)&As[wm * 64 + i * 16 + lane16][0];
      union { v16bf v; uint4v q[2]; } u;
      u.q[0] = *(const uint4v*)&arow[laneHi * 16];       // k = laneHi*16 + 0..7
      u.q[1] = *(const uint4v*)&arow[laneHi * 16 + 8];  // k = laneHi*16 + 8..15
      af[i] = u.v;
    }
#pragma unroll
    for (int j = 0; j < 2; j++) {
      const unsigned short* brow =
          (const unsigned short*)&BsT[wn * 32 + j * 16 + lane16][0] + laneHi * 16;
      union { v16bf v; uint4v q[2]; } u;
      u.q[0] = *(const uint4v*)&brow[0];                // k = laneHi*16 + 0..15
      u.q[1] = *(const uint4v*)&brow[8];
      bfr[j] = u.v;
    }
#pragma unroll
    for (int i = 0; i < 4; i++)
#pragma unroll
      for (int j = 0; j < 2; j++) acc[i][j] = WMMA_BF16(af[i], bfr[j], acc[i][j]);
    __syncthreads();
  }

#pragma unroll
  for (int i = 0; i < 4; i++)
#pragma unroll
    for (int j = 0; j < 2; j++)
#pragma unroll
      for (int r = 0; r < 8; r++) {
        const int row = bm + wm * 64 + i * 16 + r + laneHi * 8;
        const int col = bn + wn * 32 + j * 16 + lane16;
        float v = acc[i][j][r];
        if (Res) v += Res[(size_t)row * N + col];
        C[(size_t)row * N + col] = v;
      }
}

// ---------------------------------------------------------------- RoPE + head split
__global__ void rope_split(const float* __restrict__ qkv, bf16_t* __restrict__ qh,
                           bf16_t* __restrict__ kh, bf16_t* __restrict__ vh, int T) {
  const int idx = blockIdx.x * blockDim.x + threadIdx.x;  // (bt)*512 + h*32 + d
  const int d = idx & 31;
  const int h = (idx >> 5) & 15;
  const int bt = idx >> 9;
  const int t = bt & (T - 1);
  const int b = bt >> 10;  // T = 1024
  const float* row = qkv + (size_t)bt * 3072;
  const int base = h * 64 + d;
  const float q1 = row[base], q2 = row[base + 32];
  const float k1 = row[1024 + base], k2 = row[1024 + base + 32];
  const float v1 = row[2048 + base], v2 = row[2048 + base + 32];
  const float freq = (float)t * __powf(10000.f, -(float)d * (1.f / 32.f));
  float sn, cs;
  __sincosf(freq, &sn, &cs);
  const size_t o = (((size_t)b * 16 + h) * T + t) * 64 + d;
  qh[o]      = (bf16_t)(q1 * cs - q2 * sn);
  qh[o + 32] = (bf16_t)(q2 * cs + q1 * sn);
  kh[o]      = (bf16_t)(k1 * cs - k2 * sn);
  kh[o + 32] = (bf16_t)(k2 * cs + k1 * sn);
  vh[o]      = (bf16_t)v1;
  vh[o + 32] = (bf16_t)v2;
}

// ---------------------------------------------------------------- flash attention
// grid (T/64, B*NH), 128 threads; wave w owns q rows [q0+16w, q0+16w+16)
__global__ void __launch_bounds__(128)
attn_fwd(const bf16_t* __restrict__ Q, const bf16_t* __restrict__ Kh,
         const bf16_t* __restrict__ Vh, bf16_t* __restrict__ O, int T) {
  __shared__ bf16_t Ks[32][72];     // [key][dim], padded
  __shared__ bf16_t VsT[64][40];    // [dim][key], padded
  __shared__ bf16_t Ps[4][16][32];  // per-wave P relayout buffer

  const int bh = blockIdx.y;
  const int b = bh >> 4, h = bh & 15;
  const int q0 = blockIdx.x * 64;
  const bf16_t* Qp = Q + (size_t)bh * T * 64;
  const bf16_t* Kp = Kh + (size_t)bh * T * 64;
  const bf16_t* Vp = Vh + (size_t)bh * T * 64;

  const int tid = threadIdx.x, wave = tid >> 5, lane = tid & 31;
  const int lane16 = lane & 15, laneHi = (lane >> 4) & 1;
  const int qRow = q0 + wave * 16;

  v16bf qf[2];
  {
    const unsigned short* qr = (const unsigned short*)(Qp + (size_t)(qRow + lane16) * 64);
#pragma unroll
    for (int c = 0; c < 2; c++) {
      union { v16bf v; uint4v q[2]; } u;
      const int kb = c * 32 + laneHi * 16;
      u.q[0] = *(const uint4v*)&qr[kb];
      u.q[1] = *(const uint4v*)&qr[kb + 8];
      qf[c] = u.v;
    }
  }

  v8f o[4];
#pragma unroll
  for (int j = 0; j < 4; j++)
#pragma unroll
    for (int e = 0; e < 8; e++) o[j][e] = 0.f;
  float mrow[8], lrow[8];
#pragma unroll
  for (int r = 0; r < 8; r++) { mrow[r] = -3.0e38f; lrow[r] = 0.f; }

  const int ktn = (q0 + 64) >> 5;  // causal #key tiles of 32
  // staging: chunks c, c+128 -> key = c>>3, d8 = (c&7)*8  (256 chunks total)
  const int key0 = tid >> 3, d80 = (tid & 7) * 8;
  const int key1 = (tid + 128) >> 3, d81 = ((tid + 128) & 7) * 8;

  for (int kt = 0; kt < ktn; ++kt) {
    const int kb = kt * 32;
    union U8 { uint4v q; bf16_t h[8]; };
    // K: wide load, row-major store
    *(uint4v*)&Ks[key0][d80] = *(const uint4v*)(Kp + (size_t)(kb + key0) * 64 + d80);
    *(uint4v*)&Ks[key1][d81] = *(const uint4v*)(Kp + (size_t)(kb + key1) * 64 + d81);
    // V: wide load, transposed store
    U8 v0, v1;
    v0.q = *(const uint4v*)(Vp + (size_t)(kb + key0) * 64 + d80);
    v1.q = *(const uint4v*)(Vp + (size_t)(kb + key1) * 64 + d81);
#pragma unroll
    for (int j = 0; j < 8; j++) VsT[d80 + j][key0] = v0.h[j];
#pragma unroll
    for (int j = 0; j < 8; j++) VsT[d81 + j][key1] = v1.h[j];
    __syncthreads();

    // S = q @ K^T : two 16x16 tiles (keys 0-15 / 16-31), accumulated over 2 dim chunks
    v8f s0, s1;
#pragma unroll
    for (int e = 0; e < 8; e++) { s0[e] = 0.f; s1[e] = 0.f; }
#pragma unroll
    for (int c = 0; c < 2; c++) {
      const int koff = c * 32 + laneHi * 16;
      const unsigned short* k0r = (const unsigned short*)&Ks[lane16][0] + koff;
      const unsigned short* k1r = (const unsigned short*)&Ks[16 + lane16][0] + koff;
      union { v16bf v; uint4v q[2]; } u0, u1;
      u0.q[0] = *(const uint4v*)&k0r[0]; u0.q[1] = *(const uint4v*)&k0r[8];
      u1.q[0] = *(const uint4v*)&k1r[0]; u1.q[1] = *(const uint4v*)&k1r[8];
      s0 = WMMA_BF16(qf[c], u0.v, s0);
      s1 = WMMA_BF16(qf[c], u1.v, s1);
    }

    // causal mask + online softmax (row = r + 8*laneHi lives in one 16-lane half)
#pragma unroll
    for (int r = 0; r < 8; r++) {
      const int qi = qRow + laneHi * 8 + r;
      const bool m0 = (kb + lane16) <= qi;
      const bool m1 = (kb + 16 + lane16) <= qi;
      const float a  = m0 ? s0[r] * 0.125f : -3.0e38f;
      const float bb = m1 ? s1[r] * 0.125f : -3.0e38f;
      const float mt = red_max16(fmaxf(a, bb));
      const float mN = fmaxf(mrow[r], mt);
      const float scl = __expf(mrow[r] - mN);
      const float p0 = m0 ? __expf(a - mN) : 0.f;
      const float p1 = m1 ? __expf(bb - mN) : 0.f;
      lrow[r] = lrow[r] * scl + red_sum16(p0 + p1);
      mrow[r] = mN;
#pragma unroll
      for (int j = 0; j < 4; j++) o[j][r] *= scl;
      Ps[wave][laneHi * 8 + r][lane16] = (bf16_t)p0;
      Ps[wave][laneHi * 8 + r][16 + lane16] = (bf16_t)p1;
    }

    // relayout P (C layout -> A layout) through per-wave LDS
    v16bf pf;
    {
      const unsigned short* pr = (const unsigned short*)&Ps[wave][lane16][0];
      union { v16bf v; uint4v q[2]; } u;
      u.q[0] = *(const uint4v*)&pr[laneHi * 16];
      u.q[1] = *(const uint4v*)&pr[laneHi * 16 + 8];
      pf = u.v;
    }
    // O += P @ V  (4 tiles across the 64 dims)
#pragma unroll
    for (int j = 0; j < 4; j++) {
      const unsigned short* vr = (const unsigned short*)&VsT[j * 16 + lane16][0] + laneHi * 16;
      union { v16bf v; uint4v q[2]; } u;
      u.q[0] = *(const uint4v*)&vr[0];
      u.q[1] = *(const uint4v*)&vr[8];
      o[j] = WMMA_BF16(pf, u.v, o[j]);
    }
    __syncthreads();
  }

#pragma unroll
  for (int r = 0; r < 8; r++) {
    const int t = qRow + laneHi * 8 + r;
    const float inv = (lrow[r] > 0.f) ? 1.f / lrow[r] : 0.f;
    bf16_t* orow = O + ((size_t)b * T + t) * 1024 + h * 64;
#pragma unroll
    for (int j = 0; j < 4; j++) orow[j * 16 + lane16] = (bf16_t)(o[j][r] * inv);
  }
}

// ---------------------------------------------------------------- softmax over 17 (in place)
__global__ void softmax17(float* __restrict__ g, int nrows) {
  const int r = blockIdx.x * blockDim.x + threadIdx.x;
  if (r >= nrows) return;
  float* p = g + (size_t)r * 17;
  float m = p[0];
#pragma unroll
  for (int i = 1; i < 17; i++) m = fmaxf(m, p[i]);
  float e[17], s = 0.f;
#pragma unroll
  for (int i = 0; i < 17; i++) { e[i] = __expf(p[i] - m); s += e[i]; }
  const float inv = 1.f / s;
#pragma unroll
  for (int i = 0; i < 17; i++) p[i] = e[i] * inv;
}

// ---------------------------------------------------------------- linear recurrence scan
__global__ void __launch_bounds__(32)
lru_scan(const float* __restrict__ gates, const float* __restrict__ vv,
         float* __restrict__ hout, int T) {
  const int bh = blockIdx.x;
  const int b = bh >> 6, h = bh & 63;
  const int lane = threadIdx.x;
  const int i = lane >> 1, half = lane & 1;  // row i, column half
  float st[16];
#pragma unroll
  for (int j = 0; j < 16; j++) st[j] = 0.f;
  const float* gbase = gates + ((size_t)b * T) * 17408 + h * 272 + i * 17;
  const float* vbase = vv + ((size_t)b * T) * 1024 + h * 16 + i;
  float* obase = hout + ((size_t)b * T) * 1024 + h * 16 + i;
  for (int t = 0; t < T; ++t) {
    const float* g = gbase + (size_t)t * 17408;
    __builtin_prefetch(g + 17408, 0, 0);  // next timestep
    const float a0 = g[0];
    float part = 0.f;
#pragma unroll
    for (int jj = 0; jj < 8; jj++) part += g[1 + half * 8 + jj] * st[half * 8 + jj];
    const float other = __shfl_xor(part, 1, 32);
    const float vval = vbase[(size_t)t * 1024];
    const float ni = part + other + a0 * vval;  // identical on the lane pair
#pragma unroll
    for (int jj = 0; jj < 16; jj++) st[jj] = __shfl(ni, jj * 2, 32);
    if (half == 0) obase[(size_t)t * 1024] = ni;
  }
}

// ---------------------------------------------------------------- launcher
extern "C" void kernel_launch(void* const* d_in, const int* in_sizes, int n_in,
                              void* d_out, int out_size, void* d_ws, size_t ws_size,
                              hipStream_t stream) {
  (void)in_sizes; (void)n_in; (void)out_size; (void)ws_size;
  const float* x     = (const float*)d_in[0];
  const float* anw   = (const float*)d_in[1];
  const float* w_qkv = (const float*)d_in[2];
  const float* w_ao  = (const float*)d_in[3];
  const float* lnw   = (const float*)d_in[4];
  const float* w_v   = (const float*)d_in[5];
  const float* w_a   = (const float*)d_in[6];
  const float* w_op  = (const float*)d_in[7];
  float* out = (float*)d_out;

  const int B = 2, T = 1024, D = 1024, NH = 16, H = 64, Mm = 16;
  const int BT = B * T;              // 2048
  const int NA = H * Mm * (Mm + 1);  // 17408

  char* ws = (char*)d_ws;
  size_t off = 0;
  auto alloc = [&](size_t bytes) -> void* {
    void* p = ws + off;
    off += (bytes + 255) & ~(size_t)255;
    return p;
  };
  bf16_t* h_bf    = (bf16_t*)alloc((size_t)BT * D * 2);
  bf16_t* h2_bf   = (bf16_t*)alloc((size_t)BT * D * 2);
  bf16_t* wqkv_bf = (bf16_t*)alloc((size_t)D * 3 * D * 2);
  bf16_t* wao_bf  = (bf16_t*)alloc((size_t)D * D * 2);
  bf16_t* wv_bf   = (bf16_t*)alloc((size_t)D * D * 2);
  bf16_t* wa_bf   = (bf16_t*)alloc((size_t)D * NA * 2);
  bf16_t* wop_bf  = (bf16_t*)alloc((size_t)D * D * 2);
  float*  qkv     = (float*) alloc((size_t)BT * 3 * D * 4);
  bf16_t* q_bf    = (bf16_t*)alloc((size_t)BT * D * 2);
  bf16_t* k_bf    = (bf16_t*)alloc((size_t)BT * D * 2);
  bf16_t* v_bf    = (bf16_t*)alloc((size_t)BT * D * 2);
  bf16_t* o_bf    = (bf16_t*)alloc((size_t)BT * D * 2);
  float*  x2      = (float*) alloc((size_t)BT * D * 4);
  float*  vv      = (float*) alloc((size_t)BT * D * 4);
  float*  gates   = (float*) alloc((size_t)BT * NA * 4);
  float*  hout    = (float*) alloc((size_t)BT * D * 4);
  bf16_t* hout_bf = (bf16_t*)alloc((size_t)BT * D * 2);

  auto castN = [&](const float* s, bf16_t* d, size_t n) {
    cast_f32_bf16<<<(unsigned)((n + 255) / 256), 256, 0, stream>>>(s, d, (int)n);
  };

  rmsnorm_bf16<<<BT, 256, 0, stream>>>(x, anw, h_bf, D);
  castN(w_qkv, wqkv_bf, (size_t)D * 3 * D);
  castN(w_ao, wao_bf, (size_t)D * D);
  castN(w_v, wv_bf, (size_t)D * D);
  castN(w_a, wa_bf, (size_t)D * NA);
  castN(w_op, wop_bf, (size_t)D * D);

  gemm_bf16_f32<<<dim3(3 * D / 128, BT / 128), 256, 0, stream>>>(h_bf, wqkv_bf, qkv, nullptr, BT, 3 * D, D);
  rope_split<<<(BT * NH * 32) / 256, 256, 0, stream>>>(qkv, q_bf, k_bf, v_bf, T);
  attn_fwd<<<dim3(T / 64, B * NH), 128, 0, stream>>>(q_bf, k_bf, v_bf, o_bf, T);
  gemm_bf16_f32<<<dim3(D / 128, BT / 128), 256, 0, stream>>>(o_bf, wao_bf, x2, x, BT, D, D);
  rmsnorm_bf16<<<BT, 256, 0, stream>>>(x2, lnw, h2_bf, D);
  gemm_bf16_f32<<<dim3(D / 128, BT / 128), 256, 0, stream>>>(h2_bf, wv_bf, vv, nullptr, BT, D, D);
  // warm 34MB of gate weights into L2 via the Tensor Data Mover, then the big GEMM
  {
    const int rowElems8 = NA * 2 / 8;  // 4352 (8B units per bf16 row)
    const int tilesX = rowElems8 / 256;  // 17
    l2_warm_tdm<<<tilesX * (D / 16), 32, 0, stream>>>(wa_bf, rowElems8, D, tilesX);
  }
  gemm_bf16_f32<<<dim3(NA / 128, BT / 128), 256, 0, stream>>>(h2_bf, wa_bf, gates, nullptr, BT, NA, D);
  softmax17<<<(BT * H * Mm) / 256, 256, 0, stream>>>(gates, BT * H * Mm);
  lru_scan<<<B * H, 32, 0, stream>>>(gates, vv, hout, T);
  cast_f32_bf16<<<(BT * D) / 256, 256, 0, stream>>>(hout, hout_bf, BT * D);
  gemm_bf16_f32<<<dim3(D / 128, BT / 128), 256, 0, stream>>>(hout_bf, wop_bf, out, x2, BT, D, D);
}
